// NucleusMoELayer_45792941310319
// MI455X (gfx1250) — matrix operand order repaired and checked by reading, not compiled, not run on hardware
//
#include <hip/hip_runtime.h>
#include <hip/hip_bf16.h>
#include <math.h>

typedef __attribute__((ext_vector_type(16))) __bf16 v16bf;
typedef __attribute__((ext_vector_type(8)))  float  v8f;
typedef __attribute__((ext_vector_type(4)))  unsigned int u32x4;
typedef __attribute__((ext_vector_type(8)))  unsigned int u32x8;

constexpr int kBS    = 2;
constexpr int kSLEN  = 4096;
constexpr int kDIM   = 1024;
constexpr int kINNER = 2048;
constexpr int kE     = 8;
constexpr int kCAP   = 1024;
constexpr int kTOK   = kBS * kSLEN;   // 8192
constexpr int kSLOT  = kBS * kCAP;    // 2048 slots per expert

// ---------- helpers ----------

__device__ __forceinline__ unsigned short f2bf(float f) {
  unsigned int u = __float_as_uint(f);
  u = (u + 0x7FFFu + ((u >> 16) & 1u)) >> 16;  // round-to-nearest-even
  return (unsigned short)u;
}

__device__ __forceinline__ v8f wmma_bf16(v16bf a, v16bf b, v8f c) {
  // (neg_a, A, neg_b, B, c_mod, C, reuse_a, reuse_b)
  return __builtin_amdgcn_wmma_f32_16x16x32_bf16(false, a, false, b, (short)0, c,
                                                 false, false);
}

// A fragment 16x32 bf16 from row-major LDS tile (stride in ushorts, mult of 8).
// lane<16: row=row0+lane, K {0..7,16..23}; lane>=16: row=row0+lane-16, K {8..15,24..31}
__device__ __forceinline__ v16bf frag_A(const unsigned short* lds, int row0,
                                        int lane, int stride) {
  const unsigned short* p = lds + (row0 + (lane & 15)) * stride + ((lane < 16) ? 0 : 8);
  union { uint4 q[2]; v16bf v; } u;
  u.q[0] = *(const uint4*)p;
  u.q[1] = *(const uint4*)(p + 16);
  return u.v;
}

// B fragment 32x16 bf16 from column-major ([N][K]) LDS tile.
__device__ __forceinline__ v16bf frag_B(const unsigned short* lds, int col0,
                                        int lane, int stride) {
  const unsigned short* p = lds + (col0 + (lane & 15)) * stride + ((lane < 16) ? 0 : 8);
  union { uint4 q[2]; v16bf v; } u;
  u.q[0] = *(const uint4*)p;
  u.q[1] = *(const uint4*)(p + 16);
  return u.v;
}

// TDM: DMA a [tileD1 rows x tileD0 elems] bf16 tile (row stride strideElems in the
// tensor) from global into LDS at ldsAddr, inserting 16B of padding after every 64B
// row so the LDS image has a 40-ushort row stride (matches frag_A layout).
// Must be executed by a single (uniform) wave; tracked by TENSORcnt.
__device__ __forceinline__ void tdm_load_A_tile(unsigned ldsAddr,
                                                const unsigned short* gsrc,
                                                unsigned tensorD0, unsigned tensorD1,
                                                unsigned tileD0, unsigned tileD1,
                                                unsigned strideElems) {
  unsigned long long ga = (unsigned long long)(uintptr_t)gsrc;
  u32x4 g0;
  g0[0] = 1u;                                            // count=1, user mode
  g0[1] = ldsAddr;                                       // lds_addr
  g0[2] = (unsigned)ga;                                  // global_addr[31:0]
  g0[3] = (unsigned)((ga >> 32) & 0x1FFFFFFu)            // global_addr[56:32]
          | 0x80000000u;                                 // type=2 ("image")
  u32x8 g1;
  // word0: data_size=1(2B) | pad_enable | pad_interval=3(16 DW) | pad_amount=3(4 DW)
  g1[0] = 0x06D10000u;
  g1[1] = (tensorD0 & 0xFFFFu) << 16;                    // abar=0 | tensor_dim0 lo16
  g1[2] = ((tensorD0 >> 16) & 0xFFFFu) | ((tensorD1 & 0xFFFFu) << 16);
  g1[3] = ((tensorD1 >> 16) & 0xFFFFu) | (tileD0 << 16); // tile_dim0
  g1[4] = tileD1;                                        // tile_dim1 | tile_dim2=0
  g1[5] = strideElems;                                   // tensor_dim0_stride[31:0]
  g1[6] = 0u;                                            // stride0 hi | stride1 lo
  g1[7] = 0u;
  asm volatile("tensor_load_to_lds %0, %1" :: "s"(g0), "s"(g1) : "memory");
}

// ---------- K0: router (logits -> sigmoid -> affinity), zero sums ----------
__global__ __launch_bounds__(256) void router_kernel(
    const float* __restrict__ unmod, const float* __restrict__ ts,
    const float* __restrict__ Wg, float* __restrict__ aff,
    float* __restrict__ sums) {
  __shared__ float wgs[kE * 2 * kDIM];  // 64 KB
  const int tid = threadIdx.x;
  const int gtid = blockIdx.x * 256 + tid;
  if (gtid < kTOK) sums[gtid] = 0.0f;
  for (int i = tid; i < kE * 2 * kDIM; i += 256) wgs[i] = Wg[i];
  __syncthreads();

  const int lane = tid & 31, wid = tid >> 5;
  const int token = blockIdx.x * 8 + wid;       // 8 tokens per block
  const int b = token >> 12;                    // /4096
  float acc[kE];
#pragma unroll
  for (int e = 0; e < kE; ++e) acc[e] = 0.0f;

  for (int f = lane; f < 2 * kDIM; f += 32) {
    float ri = (f < kDIM) ? ts[b * kDIM + f]
                          : unmod[(size_t)token * kDIM + (f - kDIM)];
#pragma unroll
    for (int e = 0; e < kE; ++e) acc[e] += ri * wgs[e * 2 * kDIM + f];
  }
#pragma unroll
  for (int e = 0; e < kE; ++e)
    for (int off = 16; off > 0; off >>= 1) acc[e] += __shfl_down(acc[e], off);

  if (lane == 0) {
    const int s = token & (kSLEN - 1);
#pragma unroll
    for (int e = 0; e < kE; ++e)
      aff[((size_t)(b * kE + e)) * kSLEN + s] = 1.0f / (1.0f + __expf(-acc[e]));
  }
}

// ---------- K1: per-(b,e) top-CAP via bitonic sort + segment-sum ----------
__global__ __launch_bounds__(1024) void topk_kernel(
    const float* __restrict__ aff, float* __restrict__ gate,
    int* __restrict__ gidx, float* __restrict__ sums) {
  __shared__ float key[kSLEN];
  __shared__ int   val[kSLEN];
  const int be = blockIdx.x;           // b*8 + e
  const int b = be >> 3, e = be & 7;
  const float* a = aff + (size_t)be * kSLEN;
  for (int i = threadIdx.x; i < kSLEN; i += 1024) {
    key[i] = a[i];
    val[i] = b * kSLEN + i;            // global token id
  }
  __syncthreads();

  for (int k = 2; k <= kSLEN; k <<= 1) {
    for (int j = k >> 1; j > 0; j >>= 1) {
      for (int i = threadIdx.x; i < kSLEN; i += 1024) {
        int ix = i ^ j;
        if (ix > i) {
          bool desc = ((i & k) == 0);
          float ki = key[i], kx = key[ix];
          bool sw = desc ? (ki < kx) : (ki > kx);
          if (sw) {
            key[i] = kx; key[ix] = ki;
            int t = val[i]; val[i] = val[ix]; val[ix] = t;
          }
        }
      }
      __syncthreads();
    }
  }
  // expert-major output layout: [e][b][cap]
  for (int c = threadIdx.x; c < kCAP; c += 1024) {
    float g = key[c];
    int   t = val[c];
    int   o = e * kSLOT + b * kCAP + c;
    gate[o] = g;
    gidx[o] = t;
    atomicAdd(&sums[t], g);
  }
}

// ---------- K2: normalize gate weights ----------
__global__ __launch_bounds__(256) void norm_kernel(
    const float* __restrict__ gate, const int* __restrict__ gidx,
    const float* __restrict__ sums, float* __restrict__ gsc) {
  int i = blockIdx.x * 256 + threadIdx.x;
  if (i < kE * kSLOT) gsc[i] = gate[i] / (sums[gidx[i]] + 1e-12f);  // ROUTE_SCALE=1
}

// ---------- K3/K4: GEMM1 + SwiGLU (optionally gathered A) ----------
// C = X[rows] @ W (W: [1024, 4096]); H[row, n] = Cval * silu(Cgate), bf16 out.
template <bool GATHER>
__global__ __launch_bounds__(256) void gemm1_swiglu(
    const float* __restrict__ X, const int* __restrict__ gidx,
    const float* __restrict__ W, unsigned short* __restrict__ H, int Mtot) {
  constexpr int BM = 128, BN = 64, BK = 32, LD = 40;  // LD*2B = 80B (16B mult)
  __shared__ alignas(16) unsigned short As[BM * LD];
  __shared__ alignas(16) unsigned short Bs[2 * BN * LD];

  const int tid = threadIdx.x, lane = tid & 31, wid = tid >> 5;
  const int wm = wid >> 1, wn = wid & 1;  // 4 waves over M, 2 over N
  const int mBase = blockIdx.y * BM, nBase = blockIdx.x * BN, z = blockIdx.z;
  const float* Wz = W + (size_t)z * kDIM * (2 * kINNER);
  unsigned short* Hz = H + (size_t)z * (size_t)Mtot * kINNER;
  const int* idxz = GATHER ? (gidx + z * Mtot) : nullptr;

  v8f acc[2][2][2] = {};  // [im][in][val|gate]

  for (int k0 = 0; k0 < kDIM; k0 += BK) {
    for (int i = tid; i < BM * BK; i += 256) {
      int r = i >> 5, k = i & (BK - 1);
      int row = mBase + r;
      int grow = GATHER ? idxz[row] : row;
      As[r * LD + k] = f2bf(X[(size_t)grow * kDIM + (k0 + k)]);
    }
    for (int i = tid; i < BK * 2 * BN; i += 256) {
      int k = i >> 7, c = i & 127;  // c<64: value cols, c>=64: gate cols (+INNER)
      int gc = (c < BN) ? (nBase + c) : (kINNER + nBase + (c - BN));
      Bs[c * LD + k] = f2bf(Wz[(size_t)(k0 + k) * (2 * kINNER) + gc]);
    }
    __syncthreads();

    v16bf a[2], bv[2], bg[2];
#pragma unroll
    for (int i = 0; i < 2; ++i) {
      a[i]  = frag_A(As, wm * 32 + i * 16, lane, LD);
      bv[i] = frag_B(Bs, wn * 32 + i * 16, lane, LD);
      bg[i] = frag_B(Bs, BN + wn * 32 + i * 16, lane, LD);
    }
#pragma unroll
    for (int im = 0; im < 2; ++im)
#pragma unroll
      for (int in = 0; in < 2; ++in) {
        acc[im][in][0] = wmma_bf16(a[im], bv[in], acc[im][in][0]);
        acc[im][in][1] = wmma_bf16(a[im], bg[in], acc[im][in][1]);
      }
    __syncthreads();
  }

  const int cOff = lane & 15, rOff = (lane < 16) ? 0 : 8;
#pragma unroll
  for (int im = 0; im < 2; ++im)
#pragma unroll
    for (int in = 0; in < 2; ++in) {
      v8f av = acc[im][in][0], gv = acc[im][in][1];
#pragma unroll
      for (int r = 0; r < 8; ++r) {
        float g = gv[r];
        float o = av[r] * (g / (1.0f + __expf(-g)));  // a * silu(g)
        int row = mBase + wm * 32 + im * 16 + rOff + r;
        int col = nBase + wn * 32 + in * 16 + cOff;
        Hz[(size_t)row * kINNER + col] = f2bf(o);
      }
    }
}

// ---------- K5/K6: GEMM2 (optionally scaled scatter-add) ----------
// Out = H(bf16)[Mtot,2048] @ W2[2048,1024].
// A tile staged by TDM tensor_load_to_lds (double-buffered, TENSORcnt), since H is
// already bf16 row-major; B tile converted fp32->bf16 by the VALUs in parallel.
template <bool SCATTER>
__global__ __launch_bounds__(256) void gemm2_kern(
    const unsigned short* __restrict__ H, const float* __restrict__ W2,
    float* __restrict__ Out, const int* __restrict__ gidx,
    const float* __restrict__ gsc, int Mtot) {
  constexpr int BM = 128, BN = 64, BK = 32, LD = 40;
  __shared__ alignas(16) unsigned short As[2][BM * LD];
  __shared__ alignas(16) unsigned short Bs[BN * LD];

  const int tid = threadIdx.x, lane = tid & 31, wid = tid >> 5;
  const int wm = wid >> 1, wn = wid & 1;
  const int mBase = blockIdx.y * BM, nBase = blockIdx.x * BN, z = blockIdx.z;
  const unsigned short* Hz = H + (size_t)z * (size_t)Mtot * kINNER;
  const float* Wz = W2 + (size_t)z * kINNER * kDIM;
  const unsigned ldsA0 = (unsigned)(uintptr_t)(void*)&As[0][0];
  const unsigned ldsA1 = (unsigned)(uintptr_t)(void*)&As[1][0];

  v8f acc[2][2] = {};

  // prefetch first A tile via TDM (wave 0 issues; descriptor is uniform)
  if (wid == 0) {
    tdm_load_A_tile(ldsA0, Hz + (size_t)mBase * kINNER, kINNER, (unsigned)Mtot,
                    BK, BM, kINNER);
  }

  for (int k0 = 0, it = 0; k0 < kINNER; k0 += BK, ++it) {
    const unsigned short* Ac = As[it & 1];
    // B tile: fp32 -> bf16, [N][K] with padded stride
    for (int i = tid; i < BK * BN; i += 256) {
      int k = i >> 6, c = i & (BN - 1);
      Bs[c * LD + k] = f2bf(Wz[(size_t)(k0 + k) * kDIM + (nBase + c)]);
    }
    if (wid == 0) __builtin_amdgcn_s_wait_tensorcnt(0);
    __syncthreads();  // publish TDM-written A tile + Bs to all waves

    // kick off DMA of the next A tile into the other buffer (overlaps compute)
    if (wid == 0 && (k0 + BK) < kINNER) {
      tdm_load_A_tile((it & 1) ? ldsA0 : ldsA1,
                      Hz + (size_t)mBase * kINNER + (k0 + BK), kINNER,
                      (unsigned)Mtot, BK, BM, kINNER);
    }

    v16bf a[2], bb[2];
#pragma unroll
    for (int i = 0; i < 2; ++i) {
      a[i]  = frag_A(Ac, wm * 32 + i * 16, lane, LD);
      bb[i] = frag_B(Bs, wn * 32 + i * 16, lane, LD);
    }
#pragma unroll
    for (int im = 0; im < 2; ++im)
#pragma unroll
      for (int in = 0; in < 2; ++in)
        acc[im][in] = wmma_bf16(a[im], bb[in], acc[im][in]);
    __syncthreads();
  }

  const int cOff = lane & 15, rOff = (lane < 16) ? 0 : 8;
#pragma unroll
  for (int im = 0; im < 2; ++im)
#pragma unroll
    for (int in = 0; in < 2; ++in) {
#pragma unroll
      for (int r = 0; r < 8; ++r) {
        int rowL = mBase + wm * 32 + im * 16 + rOff + r;
        int col  = nBase + wn * 32 + in * 16 + cOff;
        float v = acc[im][in][r];
        if (SCATTER) {
          int   t = gidx[z * Mtot + rowL];
          float s = gsc[z * Mtot + rowL];
          atomicAdd(&Out[(size_t)t * kDIM + col], v * s);
        } else {
          Out[(size_t)rowL * kDIM + col] = v;
        }
      }
    }
}

// ---------- host ----------
extern "C" void kernel_launch(void* const* d_in, const int* in_sizes, int n_in,
                              void* d_out, int out_size, void* d_ws, size_t ws_size,
                              hipStream_t stream) {
  (void)in_sizes; (void)n_in; (void)out_size; (void)ws_size;
  const float* hidden = (const float*)d_in[0];
  const float* unmod  = (const float*)d_in[1];
  const float* ts     = (const float*)d_in[2];
  const float* Wg     = (const float*)d_in[3];
  const float* W1     = (const float*)d_in[4];
  const float* W2     = (const float*)d_in[5];
  const float* Ws1    = (const float*)d_in[6];
  const float* Ws2    = (const float*)d_in[7];
  float* out = (float*)d_out;

  char* ws = (char*)d_ws;
  size_t o = 0;
  float* aff  = (float*)(ws + o); o += (size_t)kBS * kE * kSLEN * 4;   // 256 KB
  float* gate = (float*)(ws + o); o += (size_t)kE * kSLOT * 4;         // 64 KB
  int*   gidx = (int*)  (ws + o); o += (size_t)kE * kSLOT * 4;         // 64 KB
  float* sums = (float*)(ws + o); o += (size_t)kTOK * 4;               // 32 KB
  float* gsc  = (float*)(ws + o); o += (size_t)kE * kSLOT * 4;         // 64 KB
  unsigned short* Hsh = (unsigned short*)(ws + o); o += (size_t)kTOK * kINNER * 2;        // 32 MB
  unsigned short* Hrt = (unsigned short*)(ws + o); o += (size_t)kE * kSLOT * kINNER * 2;  // 64 MB

  // 1) router + zero sums
  router_kernel<<<dim3(kTOK / 8), dim3(256), 0, stream>>>(unmod, ts, Wg, aff, sums);
  // 2) expert-choice top-k + segment sums
  topk_kernel<<<dim3(kBS * kE), dim3(1024), 0, stream>>>(aff, gate, gidx, sums);
  // 3) gate normalization
  norm_kernel<<<dim3((kE * kSLOT + 255) / 256), dim3(256), 0, stream>>>(gate, gidx, sums, gsc);
  // 4) shared expert: H = swiglu(X @ Ws1)
  gemm1_swiglu<false><<<dim3(kINNER / 64, kTOK / 128, 1), dim3(256), 0, stream>>>(
      hidden, nullptr, Ws1, Hsh, kTOK);
  // 5) routed experts: H_e = swiglu(gather(X) @ W1_e)
  gemm1_swiglu<true><<<dim3(kINNER / 64, kSLOT / 128, kE), dim3(256), 0, stream>>>(
      hidden, gidx, W1, Hrt, kSLOT);
  // 6) shared expert out-proj: writes (initializes) d_out
  gemm2_kern<false><<<dim3(kDIM / 64, kTOK / 128, 1), dim3(256), 0, stream>>>(
      Hsh, Ws2, out, nullptr, nullptr, kTOK);
  // 7) routed out-proj, scale by gate, scatter-add into d_out
  gemm2_kern<true><<<dim3(kDIM / 64, kSLOT / 128, kE), dim3(256), 0, stream>>>(
      Hrt, W2, out, gidx, gsc, kSLOT);
}
